// LocalCausalSelfAttention_36867999269435
// MI455X (gfx1250) — matrix operand order repaired and checked by reading, not compiled, not run on hardware
//
#include <hip/hip_runtime.h>

// ---------------------------------------------------------------------------
// LocalCausalSelfAttention for MI455X (gfx1250, wave32, WMMA bf16)
//   S=4096, D=1024, H=16, Dh=64, WINDOW=256
// Pipeline: f32->bf16 convert; Q/K/V projections via LDS-tiled bf16 WMMA GEMM
// (128x128 tile per 8-wave workgroup, double-buffered 64-deep K slabs in LDS);
// flash-style windowed attention (one wave per head x 16-query block);
// output projection via the same GEMM with f32 epilogue.
// ---------------------------------------------------------------------------

typedef __attribute__((ext_vector_type(16))) __bf16 v16bf;
typedef __attribute__((ext_vector_type(8)))  __bf16 v8bf;
typedef __attribute__((ext_vector_type(8)))  float  v8f;

#define SEQ    4096
#define DMODEL 1024
#define NHEAD  16
#define DHEAD  64
#define WIN    256
#define KC     64      // K-slab depth staged in LDS
#define LDSTR  72      // padded LDS row stride (bf16 elems): 36 dwords, conflict-free

__device__ __forceinline__ __bf16 to_bf16(float f) {
    unsigned u = __builtin_bit_cast(unsigned, f);
    unsigned r = u + 0x7FFFu + ((u >> 16) & 1u);   // round-to-nearest-even
    unsigned short hs = (unsigned short)(r >> 16);
    return __builtin_bit_cast(__bf16, hs);
}

__device__ __forceinline__ v8f vzero8() {
    v8f z;
#pragma unroll
    for (int i = 0; i < 8; ++i) z[i] = 0.0f;
    return z;
}

__device__ __forceinline__ v16bf cat16(v8bf lo, v8bf hi) {
    return __builtin_shufflevector(lo, hi, 0,1,2,3,4,5,6,7,8,9,10,11,12,13,14,15);
}

__device__ __forceinline__ v8f wmma_bf16(v16bf a, v16bf b, v8f c) {
    // D = A(16x32 bf16) x B(32x16 bf16) + C(16x16 f32)
    return __builtin_amdgcn_wmma_f32_16x16x32_bf16(
        false, a, false, b, (short)0, c, false, false);
}

__device__ __forceinline__ float rmax16(float v) {
#pragma unroll
    for (int m = 1; m < 16; m <<= 1) v = fmaxf(v, __shfl_xor(v, m, 32));
    return v;
}
__device__ __forceinline__ float rsum16(float v) {
#pragma unroll
    for (int m = 1; m < 16; m <<= 1) v += __shfl_xor(v, m, 32);
    return v;
}

// ---------------------------------------------------------------------------
// Kernel 1: f32 -> bf16 elementwise convert
// ---------------------------------------------------------------------------
__global__ void cvt_bf16_kernel(const float* __restrict__ src,
                                __bf16* __restrict__ dst, int n) {
    int i = blockIdx.x * blockDim.x + threadIdx.x;
    if (i < n) dst[i] = to_bf16(src[i]);
}

// ---------------------------------------------------------------------------
// Kernel 2/4: C[4096,1024] = X[4096,1024](bf16) . W[1024,1024](bf16)^T
// 256-thread block (8 waves) computes a 128x128 tile. A/B K-slabs (128x64)
// double-buffered in LDS; each wave computes 64x32 (4 M-tiles x 2 N-tiles),
// 16 WMMAs per K-slab per wave.
// MODE 0: bf16 row-major out [4096,1024]
// MODE 1: bf16 transposed out (out[col*4096 + row])
// MODE 2: f32 row-major out [4096,1024]
// ---------------------------------------------------------------------------
template <int MODE>
__global__ __launch_bounds__(256)
void gemm_bf16_kernel(const __bf16* __restrict__ X,
                      const __bf16* __restrict__ W,
                      void* __restrict__ Out) {
    const int t    = threadIdx.x;
    const int lane = t & 31;
    const int wv   = t >> 5;          // wave 0..7
    const int wm   = wv & 1;          // M sub-block (64 rows)
    const int wn   = wv >> 1;         // N sub-block (32 cols)
    const int n    = lane & 15;
    const int h    = lane >> 4;
    const int bm   = blockIdx.x * 128;
    const int bn   = blockIdx.y * 128;

    __shared__ __attribute__((aligned(16))) __bf16 As[2][128 * LDSTR];
    __shared__ __attribute__((aligned(16))) __bf16 Bs[2][128 * LDSTR];

    // ---- global fetch / LDS stage assignment: thread -> (row, 32-col half)
    const int frow = t >> 1;
    const int fcol = (t & 1) * 32;
    const __bf16* gA = X + (size_t)(bm + frow) * DMODEL + fcol;
    const __bf16* gB = W + (size_t)(bn + frow) * DMODEL + fcol;

    v8bf sa[4], sb[4];
    auto fetch = [&](int kc) {
#pragma unroll
        for (int i = 0; i < 4; ++i) {
            sa[i] = *(const v8bf*)(gA + kc * KC + i * 8);
            sb[i] = *(const v8bf*)(gB + kc * KC + i * 8);
        }
    };
    auto stage = [&](int buf) {
#pragma unroll
        for (int i = 0; i < 4; ++i) {
            *(v8bf*)&As[buf][frow * LDSTR + fcol + i * 8] = sa[i];
            *(v8bf*)&Bs[buf][frow * LDSTR + fcol + i * 8] = sb[i];
        }
    };

    v8f acc[4][2];
#pragma unroll
    for (int mt = 0; mt < 4; ++mt)
#pragma unroll
        for (int nt = 0; nt < 2; ++nt) acc[mt][nt] = vzero8();

    fetch(0);
    stage(0);
    __syncthreads();

    const int NKC = DMODEL / KC;   // 16
    for (int kc = 0; kc < NKC; ++kc) {
        const int cur = kc & 1;
        if (kc + 1 < NKC) fetch(kc + 1);   // loads in flight during compute

#pragma unroll
        for (int ks = 0; ks < 2; ++ks) {   // two k=32 steps per slab
            v16bf a[4], b[2];
#pragma unroll
            for (int mt = 0; mt < 4; ++mt) {
                const __bf16* p = &As[cur][(wm * 64 + mt * 16 + n) * LDSTR + ks * 32];
                a[mt] = cat16(*(const v8bf*)(p + 8 * h),
                              *(const v8bf*)(p + 16 + 8 * h));
            }
#pragma unroll
            for (int nt = 0; nt < 2; ++nt) {
                const __bf16* p = &Bs[cur][(wn * 32 + nt * 16 + n) * LDSTR + ks * 32];
                b[nt] = cat16(*(const v8bf*)(p + 8 * h), *(const v8bf*)(p + 16 + 8 * h));
            }
#pragma unroll
            for (int mt = 0; mt < 4; ++mt)
#pragma unroll
                for (int nt = 0; nt < 2; ++nt)
                    acc[mt][nt] = wmma_bf16(a[mt], b[nt], acc[mt][nt]);
        }

        __syncthreads();
        if (kc + 1 < NKC) {
            stage((kc + 1) & 1);
            __syncthreads();
        }
    }

#pragma unroll
    for (int mt = 0; mt < 4; ++mt)
#pragma unroll
        for (int nt = 0; nt < 2; ++nt)
#pragma unroll
            for (int r = 0; r < 8; ++r) {
                const int row = bm + wm * 64 + mt * 16 + r + 8 * h;
                const int col = bn + wn * 32 + nt * 16 + n;
                const float v = acc[mt][nt][r];
                if (MODE == 0) {
                    ((__bf16*)Out)[(size_t)row * DMODEL + col] = to_bf16(v);
                } else if (MODE == 1) {
                    ((__bf16*)Out)[(size_t)col * SEQ + row] = to_bf16(v);
                } else {
                    ((float*)Out)[(size_t)row * DMODEL + col] = v;
                }
            }
}

// ---------------------------------------------------------------------------
// Kernel 3: windowed causal attention.
// Grid (SEQ/16, NHEAD), block = 32 (one wave).
// Q,K: bf16 [S, 1024] row-major (col = head*64 + dh)
// VT : bf16 [1024, S]  (row = head*64 + dh) -> contiguous B-operand loads
// O  : bf16 [S, 1024]
// Key tiles kt in [qb-17, qb] processed as 9 pairs; tile qb-17 is fully masked.
// ---------------------------------------------------------------------------
__global__ __launch_bounds__(32)
void attn_kernel(const __bf16* __restrict__ Q,
                 const __bf16* __restrict__ K,
                 const __bf16* __restrict__ VT,
                 __bf16* __restrict__ O) {
    const int lane = threadIdx.x;
    const int n = lane & 15;
    const int h = lane >> 4;
    const int qb = blockIdx.x;
    const int head = blockIdx.y;

    __shared__ __attribute__((aligned(16))) __bf16 lds[16 * 32];  // P pair, row-major

    // Q tile as WMMA A operand (row = n, K split by lane half)
    const __bf16* qp = Q + (size_t)(qb * 16 + n) * DMODEL + head * DHEAD;
    const v16bf q0 = cat16(*(const v8bf*)(qp + 16 * h),
                           *(const v8bf*)(qp + 8 + 16 * h));        // dh 0..31
    const v16bf q1 = cat16(*(const v8bf*)(qp + 32 + 16 * h),
                           *(const v8bf*)(qp + 40 + 16 * h));       // dh 32..63

    float m_run[8], l_run[8];
#pragma unroll
    for (int r = 0; r < 8; ++r) { m_run[r] = -3.0e38f; l_run[r] = 0.0f; }
    v8f acc[4];
#pragma unroll
    for (int ct = 0; ct < 4; ++ct) acc[ct] = vzero8();

    for (int pt = 0; pt < 9; ++pt) {
        const int kt0 = qb - 17 + 2 * pt;

        // ---- scores for the two 16-key tiles ----
        float st[2][8];
#pragma unroll
        for (int t = 0; t < 2; ++t) {
            const int kt = kt0 + t;
            const int keyrow = kt * 16 + n;                 // B column n -> key index
            const int kr = keyrow < 0 ? 0 : keyrow;         // clamp (masked anyway)
            const __bf16* kp = K + (size_t)kr * DMODEL + head * DHEAD;
            v16bf b0 = *(const v16bf*)(kp + 16 * h);        // dh 0..31 split by half
            v16bf b1 = *(const v16bf*)(kp + 32 + 16 * h);   // dh 32..63
            v8f c = vzero8();
            c = wmma_bf16(q0, b0, c);
            c = wmma_bf16(q1, b1, c);
#pragma unroll
            for (int r = 0; r < 8; ++r) {
                const int i = qb * 16 + r + 8 * h;          // query index (C-layout row)
                const int j = keyrow;                       // key index  (C-layout col)
                const bool ok = (j >= 0) && (j <= i) && (j > i - WIN);
                st[t][r] = ok ? c[r] * 0.125f : -3.0e38f;   // scale = 1/sqrt(64)
            }
        }

        // ---- online softmax update ----
        float resc[8], pv[2][8];
#pragma unroll
        for (int r = 0; r < 8; ++r) {
            float mc = rmax16(fmaxf(st[0][r], st[1][r]));
            float mn = fmaxf(m_run[r], mc);
            resc[r] = (mn == m_run[r]) ? 1.0f : __expf(m_run[r] - mn);
            float p0 = (st[0][r] > -1.0e38f) ? __expf(st[0][r] - mn) : 0.0f;
            float p1 = (st[1][r] > -1.0e38f) ? __expf(st[1][r] - mn) : 0.0f;
            pv[0][r] = p0; pv[1][r] = p1;
            l_run[r] = l_run[r] * resc[r] + rsum16(p0 + p1);
            m_run[r] = mn;
        }
#pragma unroll
        for (int ct = 0; ct < 4; ++ct)
#pragma unroll
            for (int r = 0; r < 8; ++r) acc[ct][r] *= resc[r];

        // ---- P: C-layout -> row-major LDS -> A-operand layout ----
#pragma unroll
        for (int t = 0; t < 2; ++t)
#pragma unroll
            for (int r = 0; r < 8; ++r)
                lds[(r + 8 * h) * 32 + 16 * t + n] = to_bf16(pv[t][r]);
        __syncthreads();
        const v16bf pa = cat16(*(const v8bf*)&lds[n * 32 + 16 * h],
                               *(const v8bf*)&lds[n * 32 + 8 + 16 * h]);
        __syncthreads();

        // ---- out += P(16x32) x V(32x16) per 16-wide dh column tile ----
        int keybase = kt0 * 16 + 16 * h;                    // lane-half K offset
        keybase = keybase < 0 ? 0 : keybase;                // masked rows have P==0
#pragma unroll
        for (int ct = 0; ct < 4; ++ct) {
            const __bf16* vp = VT + (size_t)(head * DHEAD + ct * 16 + n) * SEQ + keybase;
            v16bf vb = *(const v16bf*)vp;
            acc[ct] = wmma_bf16(pa, vb, acc[ct]);
        }
    }

    // ---- normalize + store bf16 [S, 1024] ----
#pragma unroll
    for (int ct = 0; ct < 4; ++ct)
#pragma unroll
        for (int r = 0; r < 8; ++r) {
            const int row = qb * 16 + r + 8 * h;
            const int col = head * DHEAD + ct * 16 + n;
            O[(size_t)row * DMODEL + col] = to_bf16(acc[ct][r] / l_run[r]);
        }
}

// ---------------------------------------------------------------------------
// Host launcher
// ---------------------------------------------------------------------------
extern "C" void kernel_launch(void* const* d_in, const int* in_sizes, int n_in,
                              void* d_out, int out_size, void* d_ws, size_t ws_size,
                              hipStream_t stream) {
    const float* hidden = (const float*)d_in[0];
    const float* Wq = (const float*)d_in[1];
    const float* Wk = (const float*)d_in[2];
    const float* Wv = (const float*)d_in[3];
    const float* Wo = (const float*)d_in[4];
    float* out = (float*)d_out;

    char* w = (char*)d_ws;
    const size_t MB = 1u << 20;
    __bf16* Xb  = (__bf16*)(w + 0 * MB);     // [4096,1024]  8 MB
    __bf16* Wqb = (__bf16*)(w + 8 * MB);     // [1024,1024]  2 MB
    __bf16* Wkb = (__bf16*)(w + 10 * MB);
    __bf16* Wvb = (__bf16*)(w + 12 * MB);
    __bf16* Wob = (__bf16*)(w + 14 * MB);
    __bf16* Qb  = (__bf16*)(w + 16 * MB);    // [4096,1024]  8 MB
    __bf16* Kb  = (__bf16*)(w + 24 * MB);
    __bf16* VTb = (__bf16*)(w + 32 * MB);    // [1024,4096]  8 MB
    __bf16* Ab  = (__bf16*)(w + 40 * MB);    // attention out bf16, 8 MB

    const int nH = SEQ * DMODEL;      // 4M
    const int nW = DMODEL * DMODEL;   // 1M
    cvt_bf16_kernel<<<(nH + 255) / 256, 256, 0, stream>>>(hidden, Xb, nH);
    cvt_bf16_kernel<<<(nW + 255) / 256, 256, 0, stream>>>(Wq, Wqb, nW);
    cvt_bf16_kernel<<<(nW + 255) / 256, 256, 0, stream>>>(Wk, Wkb, nW);
    cvt_bf16_kernel<<<(nW + 255) / 256, 256, 0, stream>>>(Wv, Wvb, nW);
    cvt_bf16_kernel<<<(nW + 255) / 256, 256, 0, stream>>>(Wo, Wob, nW);

    dim3 ggrid(SEQ / 128, DMODEL / 128);   // 32 x 8 workgroups, 8 waves each
    gemm_bf16_kernel<0><<<ggrid, 256, 0, stream>>>(Xb, Wqb, (void*)Qb);
    gemm_bf16_kernel<0><<<ggrid, 256, 0, stream>>>(Xb, Wkb, (void*)Kb);
    gemm_bf16_kernel<1><<<ggrid, 256, 0, stream>>>(Xb, Wvb, (void*)VTb);

    dim3 agrid(SEQ / 16, NHEAD);           // 256 x 16 single-wave blocks
    attn_kernel<<<agrid, 32, 0, stream>>>(Qb, Kb, VTb, Ab);

    gemm_bf16_kernel<2><<<ggrid, 256, 0, stream>>>(Ab, Wob, (void*)out);
}
